// GraphModel_8254927143009
// MI455X (gfx1250) — matrix operand order repaired and checked by reading, not compiled, not run on hardware
//
#include <hip/hip_runtime.h>
#include <math.h>

#define NNODES   100000
#define HID      128
#define H3       384
#define NTYPES   3
#define NEDGES   300000
#define NSUBTOK  200000

typedef __attribute__((ext_vector_type(16))) __bf16 v16bf;
typedef __attribute__((ext_vector_type(4)))  __bf16 v4bf;
typedef __attribute__((ext_vector_type(8)))  float  v8f;

__device__ __forceinline__ v4bf cvt4(float4 v) {
    v4bf r;
    r[0] = (__bf16)v.x; r[1] = (__bf16)v.y;
    r[2] = (__bf16)v.z; r[3] = (__bf16)v.w;
    return r;
}

// ---------------------------------------------------------------- zero
__global__ void zero_kernel(float4* __restrict__ p, int n4) {
    int i = blockIdx.x * blockDim.x + threadIdx.x;
    if (i < n4) p[i] = make_float4(0.f, 0.f, 0.f, 0.f);
}

// ---------------------------------------------- weight pre-pack (fp32 -> bf16)
// P[((kc*NT + nt)*32 + lane)*16 + e] = W[kc*32 + ((lane>=16)?8:0) + (e&7) + ((e>=8)?16:0)][nt*16 + (lane&15)]
__global__ void pack_kernel(const float* __restrict__ W, __bf16* __restrict__ P,
                            int K, int N) {
    int flat = blockIdx.x * blockDim.x + threadIdx.x;
    if (flat >= K * N) return;
    int ntiles = N >> 4;
    int e    = flat & 15;
    int lane = (flat >> 4) & 31;
    int nt   = (flat >> 9) % ntiles;
    int kc   = flat / (ntiles << 9);
    int k = kc * 32 + ((lane >> 4) << 3) + (e & 7) + ((e >> 3) << 4);
    int n = nt * 16 + (lane & 15);
    P[flat] = (__bf16)W[k * N + n];
}

// ------------------------------------------------- embedding scatter-sum
__global__ void embed_kernel(const int* __restrict__ ids,
                             const int* __restrict__ locs,
                             const float* __restrict__ emb,
                             float* __restrict__ h0, int S) {
    int idx = blockIdx.x * blockDim.x + threadIdx.x;
    int s = idx >> 5;
    if (s >= S) return;
    int c = (idx & 31) * 4;
    int id = ids[s], loc = locs[s];
    float4 e = *(const float4*)(emb + id * HID + c);
    float* o = h0 + loc * HID + c;
    atomicAdd(o + 0, e.x); atomicAdd(o + 1, e.y);
    atomicAdd(o + 2, e.z); atomicAdd(o + 3, e.w);
}

// -------------------- fused gather -> [16,128]x[128,128] WMMA -> scatter
// block = 256 threads (8 waves), 16 edges; wave w owns output cols [16w,16w+16)
__global__ void __launch_bounds__(256)
msg_kernel(const float* __restrict__ states, const __bf16* __restrict__ Wp,
           const float* __restrict__ bias, const int* __restrict__ esrc,
           const int* __restrict__ etgt, float* __restrict__ agg, int E) {
    __shared__ __bf16 As[16][HID];
    __shared__ int srcRow[16], tgtRow[16];
    const int tid  = threadIdx.x;
    const int wave = tid >> 5, lane = tid & 31;
    const int tileBase = blockIdx.x * 16;

    if (tid < 16) {
        int e = tileBase + tid;
        if (e >= E) e = E - 1;
        srcRow[tid] = esrc[e];
        tgtRow[tid] = etgt[e];
    }
    __syncthreads();

    #pragma unroll
    for (int i = 0; i < 2; ++i) {               // 16x128 f32 -> bf16 LDS (vec4)
        int flat = tid * 2 + i;                 // 0..511 quads
        int row = flat >> 5, col4 = (flat & 31) * 4;
        float4 v = *(const float4*)(states + srcRow[row] * HID + col4);
        *(v4bf*)&As[row][col4] = cvt4(v);
    }
    __syncthreads();

    const int m    = lane & 15;
    const int kbA  = (lane < 16) ? 0 : 8;       // A: 16-bit 16x32 layout
    const int ncol = wave * 16 + (lane & 15);
    v8f acc = {};
    #pragma unroll
    for (int kc = 0; kc < 4; ++kc) {            // K = 128 in 4 chunks of 32
        v16bf a;
        #pragma unroll
        for (int j = 0; j < 8; ++j) {
            a[j]     = As[m][kc * 32 + kbA + j];
            a[8 + j] = As[m][kc * 32 + kbA + 16 + j];
        }
        v16bf b = *(const v16bf*)(Wp + ((kc * 8 + wave) * 32 + lane) * 16);
        acc = __builtin_amdgcn_wmma_f32_16x16x32_bf16(
                  false, a, false, b, (short)0, acc, false, false);
    }
    const float bval = bias[ncol];
    const int mbase = (lane < 16) ? 0 : 8;
    if (tileBase + 16 <= E) {                   // full tile: uniform fast path
        #pragma unroll
        for (int r = 0; r < 8; ++r)
            atomicAdd(agg + tgtRow[mbase + r] * HID + ncol, acc[r] + bval);
    } else {                                    // generic tail
        #pragma unroll
        for (int r = 0; r < 8; ++r)
            if (tileBase + mbase + r < E)
                atomicAdd(agg + tgtRow[mbase + r] * HID + ncol, acc[r] + bval);
    }
}

// ------------- fully fused GRU: mx = x@kernel, mh = h@rec, gates, output
// block = 16 nodes; wave w owns 3 N-tiles of the 384-wide mx AND mh
__global__ void __launch_bounds__(256)
gru_kernel(const float* __restrict__ x1, const float* __restrict__ x2,
           const float* __restrict__ h,  const __bf16* __restrict__ kernP,
           const __bf16* __restrict__ rkernP, const float* __restrict__ bias,
           float* __restrict__ out, int nNodes, int xChunks) {
    __shared__ __bf16 Xs[16][32];
    __shared__ __bf16 Hs[16][32];
    __shared__ float  Mx[16][H3];
    __shared__ float  Mh[16][H3];
    const int tid  = threadIdx.x;
    const int wave = tid >> 5, lane = tid & 31;
    const int nodeBase = blockIdx.x * 16;
    const int m   = lane & 15;
    const int kbA = (lane < 16) ? 0 : 8;

    v8f accx[3] = {};
    v8f acch[3] = {};

    for (int kc = 0; kc < xChunks; ++kc) {      // K chunks of 32
        const float* xsrc = (kc < 4) ? x1 : x2; // layer1: [h0 ; agg] concat
        const int krow = (kc & 3) * 32;
        __syncthreads();
        if (tid < 128) {                        // stage 16x32 tiles via float4
            int row = tid >> 3, col4 = (tid & 7) * 4;
            int node = nodeBase + row; if (node >= nNodes) node = nNodes - 1;
            float4 xv = *(const float4*)(xsrc + node * HID + krow + col4);
            *(v4bf*)&Xs[row][col4] = cvt4(xv);
            if (kc < 4) {
                float4 hv = *(const float4*)(h + node * HID + krow + col4);
                *(v4bf*)&Hs[row][col4] = cvt4(hv);
            }
        }
        __syncthreads();

        v16bf a;
        #pragma unroll
        for (int j = 0; j < 8; ++j) {
            a[j]     = Xs[m][kbA + j];
            a[8 + j] = Xs[m][kbA + 16 + j];
        }
        #pragma unroll
        for (int s = 0; s < 3; ++s) {
            int nt = wave * 3 + s;              // 24 N-tiles cover 384 cols
            v16bf b = *(const v16bf*)(kernP + ((kc * 24 + nt) * 32 + lane) * 16);
            accx[s] = __builtin_amdgcn_wmma_f32_16x16x32_bf16(
                          false, a, false, b, (short)0, accx[s], false, false);
        }
        if (kc < 4) {                            // mh: K fixed at 128
            v16bf ah;
            #pragma unroll
            for (int j = 0; j < 8; ++j) {
                ah[j]     = Hs[m][kbA + j];
                ah[8 + j] = Hs[m][kbA + 16 + j];
            }
            #pragma unroll
            for (int s = 0; s < 3; ++s) {
                int nt = wave * 3 + s;
                v16bf b = *(const v16bf*)(rkernP + ((kc * 24 + nt) * 32 + lane) * 16);
                acch[s] = __builtin_amdgcn_wmma_f32_16x16x32_bf16(
                              false, ah, false, b, (short)0, acch[s], false, false);
            }
        }
    }
    __syncthreads();
    #pragma unroll
    for (int s = 0; s < 3; ++s) {                // spill accs to LDS
        int ncol = (wave * 3 + s) * 16 + (lane & 15);
        int mbase = (lane < 16) ? 0 : 8;
        #pragma unroll
        for (int r = 0; r < 8; ++r) {
            Mx[mbase + r][ncol] = accx[s][r];
            Mh[mbase + r][ncol] = acch[s][r];
        }
    }
    __syncthreads();
    const bool full = (nodeBase + 16 <= nNodes); // uniform
    #pragma unroll
    for (int i = 0; i < 8; ++i) {                // gates (keras reset_after)
        int idx = tid + i * 256;
        int mm = idx >> 7, d = idx & 127;
        int node = nodeBase + mm;
        if (full || node < nNodes) {
            float xz = Mx[mm][d]           + bias[d];
            float xr = Mx[mm][HID + d]     + bias[HID + d];
            float xh = Mx[mm][2 * HID + d] + bias[2 * HID + d];
            float rz = Mh[mm][d]           + bias[H3 + d];
            float rr = Mh[mm][HID + d]     + bias[H3 + HID + d];
            float rh = Mh[mm][2 * HID + d] + bias[H3 + 2 * HID + d];
            float z  = 1.0f / (1.0f + expf(-(xz + rz)));
            float rg = 1.0f / (1.0f + expf(-(xr + rr)));
            float hh = tanhf(xh + rg * rh);
            float hv = h[node * HID + d];
            out[node * HID + d] = z * hv + (1.0f - z) * hh;
        }
    }
}

// ---------------------------------------------------------------- launch
extern "C" void kernel_launch(void* const* d_in, const int* in_sizes, int n_in,
                              void* d_out, int out_size, void* d_ws, size_t ws_size,
                              hipStream_t stream) {
    (void)in_sizes; (void)n_in; (void)out_size; (void)ws_size;
    const int*   node_ids  = (const int*)d_in[0];
    const int*   node_locs = (const int*)d_in[1];
    const int*   esrc      = (const int*)d_in[2];   // [3][E]
    const int*   etgt      = (const int*)d_in[3];   // [3][E]
    const float* emb       = (const float*)d_in[4];
    const float* tw        = (const float*)d_in[5]; // [2][3][128][128]
    const float* tb        = (const float*)d_in[6]; // [2][3][128]
    const float* k0        = (const float*)d_in[7];
    const float* rk0       = (const float*)d_in[8];
    const float* b0        = (const float*)d_in[9];
    const float* k1        = (const float*)d_in[10];
    const float* rk1       = (const float*)d_in[11];
    const float* b1        = (const float*)d_in[12];
    float* out = (float*)d_out;

    const long NH = (long)NNODES * HID;
    float* h0  = (float*)d_ws;
    float* sA  = h0 + NH;
    float* sB  = sA + NH;
    float* agg = sB + NH;
    // bf16 packed weights after the fp32 state buffers (32B-aligned offsets)
    __bf16* twP  = (__bf16*)(agg + NH);             // 6*128*128
    __bf16* k0P  = twP  + 6L * HID * HID;           // 128*384
    __bf16* rk0P = k0P  + (long)HID * H3;           // 128*384
    __bf16* k1P  = rk0P + (long)HID * H3;           // 256*384
    __bf16* rk1P = k1P  + 2L * HID * H3;            // 128*384

    const int n4    = (int)(NH / 4);
    const int zgrid = (n4 + 255) / 256;
    const int mgrid = (NEDGES + 15) / 16;
    const int ggrid = (NNODES + 15) / 16;

    // one-time weight packing (deterministic per launch, ~1MB total)
    pack_kernel<<<(6 * HID * HID + 255) / 256, 256, 0, stream>>>(tw,  twP,  6 * HID, HID);
    pack_kernel<<<(HID * H3 + 255) / 256,     256, 0, stream>>>(k0,  k0P,  HID,     H3);
    pack_kernel<<<(HID * H3 + 255) / 256,     256, 0, stream>>>(rk0, rk0P, HID,     H3);
    pack_kernel<<<(2 * HID * H3 + 255) / 256, 256, 0, stream>>>(k1,  k1P,  2 * HID, H3);
    pack_kernel<<<(HID * H3 + 255) / 256,     256, 0, stream>>>(rk1, rk1P, HID,     H3);

    zero_kernel<<<zgrid, 256, 0, stream>>>((float4*)h0, n4);
    {
        long tot = (long)NSUBTOK * 32;
        embed_kernel<<<(int)((tot + 255) / 256), 256, 0, stream>>>(
            node_ids, node_locs, emb, h0, NSUBTOK);
    }

    auto propagate = [&](const float* st, int layer) {
        zero_kernel<<<zgrid, 256, 0, stream>>>((float4*)agg, n4);
        for (int t = 0; t < NTYPES; ++t) {
            msg_kernel<<<mgrid, 256, 0, stream>>>(
                st, twP + (long)(layer * NTYPES + t) * HID * HID,
                tb + (long)(layer * NTYPES + t) * HID,
                esrc + (long)t * NEDGES, etgt + (long)t * NEDGES, agg, NEDGES);
        }
    };

    // layer 0, step 0 (states = h0)
    propagate(h0, 0);
    gru_kernel<<<ggrid, 256, 0, stream>>>(agg, agg, h0, k0P, rk0P, b0, sA, NNODES, 4);
    // layer 0, step 1
    propagate(sA, 0);
    gru_kernel<<<ggrid, 256, 0, stream>>>(agg, agg, sA, k0P, rk0P, b0, sB, NNODES, 4);
    // layer 1 (residual: mx = h0 @ k1[0:128] + agg @ k1[128:256])
    propagate(sB, 1);
    gru_kernel<<<ggrid, 256, 0, stream>>>(h0, agg, sB, k1P, rk1P, b1, out, NNODES, 8);
}
